// GroupedQueryAttention_56891136803348
// MI455X (gfx1250) — matrix operand order repaired and checked by reading, not compiled, not run on hardware
//
#include <hip/hip_runtime.h>
#include <hip/hip_bf16.h>
#include <math.h>

// ---------------------------------------------------------------------------
// GQA forward for MI455X / gfx1250.  All GEMMs + attention use
// v_wmma_f32_16x16x32_bf16 (bf16 inputs, fp32 accumulation), softmax in fp32.
// GEMM waves compute 32x64 macro-tiles (8 WMMA / K-step, no spills);
// attention waves process 32 queries sharing K/V fragment loads.
// __launch_bounds__(128,1) relaxes the VGPR budget (1 wave/EU floor).
// ---------------------------------------------------------------------------

#define D_MODEL 2048
#define H_Q     32
#define H_KV    8
#define D_HEAD  64
#define BATCH   2
#define SEQ     2048
#define ROWS    (BATCH * SEQ)                  // 4096
#define N_QKV   (D_MODEL + 2 * H_KV * D_HEAD)  // 3072

typedef __attribute__((ext_vector_type(16))) __bf16        bfrag_t;
typedef __attribute__((ext_vector_type(8)))  float         v8f;
typedef __attribute__((ext_vector_type(4)))  unsigned int  v4u;

union BF16Frag {
    bfrag_t v;
    v4u     q[2];
    unsigned short s[16];
};

__device__ __forceinline__ unsigned short f2bf(float f) {
    unsigned int u = __float_as_uint(f);
    unsigned int r = u + 0x7FFFu + ((u >> 16) & 1u);   // round-to-nearest-even
    return (unsigned short)(r >> 16);
}

// A-fragment (16x32, MxK) from a row-major bf16 matrix.
// Layout: lanes 0-15 hold M=lane with K k0..k0+15 sequential;
//         lanes 16-31 hold M=lane-16 with K k0+16..k0+31.
__device__ __forceinline__ bfrag_t load_a_frag(const unsigned short* base, int ld,
                                               int m0, int k0, int lane) {
    int half = lane >> 4, lr = lane & 15;
    const unsigned short* p = base + (size_t)(m0 + lr) * ld + k0 + (half ? 16 : 0);
    BF16Frag f;
    f.q[0] = *(const v4u*)(p);
    f.q[1] = *(const v4u*)(p + 8);
    return f.v;
}

// B-fragment (32x16, KxN) loaded from B^T stored row-major (N x K).
// Layout (contiguous k, same as A): lanes 0-15 hold N=lane with K k0..k0+15 sequential;
//             lanes 16-31 hold N=lane-16 with K k0+16..k0+31.
__device__ __forceinline__ bfrag_t load_bt_frag(const unsigned short* bt, int ld,
                                                int n0, int k0, int lane) {
    int half = lane >> 4, lr = lane & 15;
    const unsigned short* p = bt + (size_t)(n0 + lr) * ld + k0 + (half ? 16 : 0);
    BF16Frag f;
    f.q[0] = *(const v4u*)(p);
    f.q[1] = *(const v4u*)(p + 8);
    return f.v;
}

__device__ __forceinline__ v8f wmma_bf16(bfrag_t a, bfrag_t b, v8f c) {
    return __builtin_amdgcn_wmma_f32_16x16x32_bf16(
        /*neg_a=*/false, a, /*neg_b=*/false, b,
        /*c_mod=*/(short)0, c, /*reuse_a=*/false, /*reuse_b=*/false);
}

// ---------------------------------------------------------------------------
// Conversion kernels (fp32 -> bf16, weights transposed for B-fragment loads)
// ---------------------------------------------------------------------------
__global__ void cvt_x_kernel(const float* __restrict__ x,
                             unsigned short* __restrict__ xb, int n) {
    int i = blockIdx.x * blockDim.x + threadIdx.x;
    if (i < n) xb[i] = f2bf(x[i]);
}

// wt[c][k] = concat(Wq,Wk,Wv)[k][c], c in [0,3072), k in [0,2048)
__global__ void cvt_wqkv_kernel(const float* __restrict__ wq,
                                const float* __restrict__ wk,
                                const float* __restrict__ wv,
                                unsigned short* __restrict__ wt) {
    int i = blockIdx.x * blockDim.x + threadIdx.x;
    if (i >= N_QKV * D_MODEL) return;
    int c = i >> 11, k = i & 2047;
    float v;
    if (c < D_MODEL)            v = wq[(size_t)k * D_MODEL + c];
    else if (c < D_MODEL + 512) v = wk[(size_t)k * 512 + (c - D_MODEL)];
    else                        v = wv[(size_t)k * 512 + (c - D_MODEL - 512)];
    wt[i] = f2bf(v);   // i == c*2048 + k
}

__global__ void cvt_wo_kernel(const float* __restrict__ wo,
                              unsigned short* __restrict__ wot) {
    int i = blockIdx.x * blockDim.x + threadIdx.x;
    if (i >= D_MODEL * D_MODEL) return;
    int c = i >> 11, k = i & 2047;
    wot[i] = f2bf(wo[(size_t)k * D_MODEL + c]);
}

// ---------------------------------------------------------------------------
// Fused QKV projection GEMM: Y(4096 x 3072) = x_bf @ [Wq|Wk|Wv]
// Each wave computes a 32x64 C macro-tile (2x4 WMMA tiles): per K-step, load
// 4 B-fragments once and stream 2 A-fragments against them (8 WMMA / step).
// Output scattered into per-head layouts:
//   Q: [b*32+h][l][64]   K: [b*8+h][l][64]   V^T: [b*8+h][64][2048]
// ---------------------------------------------------------------------------
__global__ void __launch_bounds__(128, 1)
qkv_gemm_kernel(const unsigned short* __restrict__ xbf,
                const unsigned short* __restrict__ wt,
                unsigned short* __restrict__ qout,
                unsigned short* __restrict__ kout,
                unsigned short* __restrict__ vtout) {
    int lane = threadIdx.x & 31;
    int wave = threadIdx.x >> 5;
    int m0 = blockIdx.x * 32;                 // 128 row macro-tiles
    int n0 = (blockIdx.y * 4 + wave) * 64;    // 48 col macro-tiles

    v8f acc[2][4] = {};
    for (int k0 = 0; k0 < D_MODEL; k0 += 32) {
        bfrag_t bf[4];
        #pragma unroll
        for (int j = 0; j < 4; ++j)
            bf[j] = load_bt_frag(wt, D_MODEL, n0 + 16 * j, k0, lane);
        if (k0 + 64 <= D_MODEL) {
            __builtin_prefetch(xbf + (size_t)(m0 + (lane & 15)) * D_MODEL + k0 + 32, 0, 1);
            __builtin_prefetch(wt  + (size_t)(n0 + (lane & 15)) * D_MODEL + k0 + 32, 0, 1);
        }
        #pragma unroll
        for (int i = 0; i < 2; ++i) {
            bfrag_t af = load_a_frag(xbf, D_MODEL, m0 + 16 * i, k0, lane);
            #pragma unroll
            for (int j = 0; j < 4; ++j)
                acc[i][j] = wmma_bf16(af, bf[j], acc[i][j]);
        }
    }

    int half = lane >> 4, lr = lane & 15;
    #pragma unroll
    for (int i = 0; i < 2; ++i) {
        #pragma unroll
        for (int j = 0; j < 4; ++j) {
            int col = n0 + 16 * j + lr;
            #pragma unroll
            for (int v = 0; v < 8; ++v) {
                int row = m0 + 16 * i + v + half * 8;   // 0..4095
                int bb = row >> 11, l = row & 2047;
                unsigned short bfv = f2bf(acc[i][j][v]);
                if (col < D_MODEL) {                    // Q
                    int h = col >> 6, d = col & 63;
                    qout[((size_t)(bb * H_Q + h) * SEQ + l) * D_HEAD + d] = bfv;
                } else if (col < D_MODEL + 512) {       // K
                    int c = col - D_MODEL, h = c >> 6, d = c & 63;
                    kout[((size_t)(bb * H_KV + h) * SEQ + l) * D_HEAD + d] = bfv;
                } else {                                // V (transposed store)
                    int c = col - D_MODEL - 512, h = c >> 6, d = c & 63;
                    vtout[((size_t)(bb * H_KV + h) * D_HEAD + d) * SEQ + l] = bfv;
                }
            }
        }
    }
}

// ---------------------------------------------------------------------------
// Online-softmax update for one 16-query tile over a 32-key block.
// s0/s1 are the two 16x16 S^T score tiles (already scaled).  Updates running
// max/sum and rescales the 4 O^T accumulators; returns the P B-fragment.
// ---------------------------------------------------------------------------
__device__ __forceinline__ bfrag_t softmax_update(v8f s0, v8f s1,
                                                  float& run_m, float& run_s,
                                                  v8f (&o)[4], bool hi) {
    float mx = -1e30f;
    #pragma unroll
    for (int v = 0; v < 8; ++v) {
        mx = fmaxf(mx, s0[v]);
        mx = fmaxf(mx, s1[v]);
    }
    mx = fmaxf(mx, __shfl_xor(mx, 16, 32));
    float nm = fmaxf(run_m, mx);
    float corr = __expf(run_m - nm);

    float p0[8], p1[8], lsum = 0.0f;
    #pragma unroll
    for (int v = 0; v < 8; ++v) {
        p0[v] = __expf(s0[v] - nm);
        p1[v] = __expf(s1[v] - nm);
        lsum += p0[v] + p1[v];
    }
    lsum += __shfl_xor(lsum, 16, 32);
    run_s = run_s * corr + lsum;
    run_m = nm;

    o[0] = o[0] * corr; o[1] = o[1] * corr;
    o[2] = o[2] * corr; o[3] = o[3] * corr;

    // Assemble P (32 keys x 16 queries) B-fragment:
    //   lane<16  : K elems 0..7 = own p0 (keys 0-7),  8..15 = partner p0 (8-15)
    //   lane>=16 : K elems 0..7 = partner p1 (16-23), 8..15 = own p1   (24-31)
    BF16Frag pf;
    #pragma unroll
    for (int v = 0; v < 8; ++v) {
        float p0x = __shfl_xor(p0[v], 16, 32);
        float p1x = __shfl_xor(p1[v], 16, 32);
        pf.s[v]     = f2bf(hi ? p1x   : p0[v]);
        pf.s[v + 8] = f2bf(hi ? p1[v] : p0x);
    }
    return pf.v;
}

// ---------------------------------------------------------------------------
// Flash attention: one wave owns one (b, hq, 32-query pair of tiles).
// S^T = K @ Q^T so each lane owns one query column; K-tile and V^T-tile
// fragments are loaded once per 32-key step and reused by both query tiles
// (16 WMMA per step).  O^T = V^T @ P accumulated over 4 head-dim tiles.
// ---------------------------------------------------------------------------
__global__ void __launch_bounds__(128, 1)
attn_kernel(const unsigned short* __restrict__ qb,
            const unsigned short* __restrict__ kb,
            const unsigned short* __restrict__ vtb,
            unsigned short* __restrict__ attnb) {
    int lane = threadIdx.x & 31;
    int wave = threadIdx.x >> 5;
    int qt = blockIdx.x * 4 + wave;           // 0..63, 32 queries each
    int bh = blockIdx.y;                      // 0..63
    int b = bh >> 5, hq = bh & 31, hkv = hq >> 2;

    const unsigned short* qp = qb  + (size_t)(b * H_Q  + hq)  * SEQ * D_HEAD;
    const unsigned short* kp = kb  + (size_t)(b * H_KV + hkv) * SEQ * D_HEAD;
    const unsigned short* vp = vtb + (size_t)(b * H_KV + hkv) * D_HEAD * SEQ;
    int q0 = qt * 32;

    // Q^T B-fragments for both query tiles, head dims [0,32) / [32,64).
    bfrag_t qfA0 = load_bt_frag(qp, D_HEAD, q0,      0,  lane);
    bfrag_t qfA1 = load_bt_frag(qp, D_HEAD, q0,      32, lane);
    bfrag_t qfB0 = load_bt_frag(qp, D_HEAD, q0 + 16, 0,  lane);
    bfrag_t qfB1 = load_bt_frag(qp, D_HEAD, q0 + 16, 32, lane);

    const float scale = 0.125f;               // 1/sqrt(64)
    float run_mA = -1e30f, run_sA = 0.0f;
    float run_mB = -1e30f, run_sB = 0.0f;
    v8f oA[4] = {}, oB[4] = {};
    bool hi = (lane & 16) != 0;

    #pragma unroll 1
    for (int ks = 0; ks < SEQ; ks += 32) {
        // K A-fragments for key rows [ks,ks+16) and [ks+16,ks+32), shared by
        // both query tiles.
        bfrag_t ka0 = load_a_frag(kp, D_HEAD, ks,      0,  lane);
        bfrag_t ka1 = load_a_frag(kp, D_HEAD, ks,      32, lane);
        bfrag_t kb0 = load_a_frag(kp, D_HEAD, ks + 16, 0,  lane);
        bfrag_t kb1 = load_a_frag(kp, D_HEAD, ks + 16, 32, lane);

        v8f sA0 = {}, sA1 = {}, sB0 = {}, sB1 = {};
        sA0 = wmma_bf16(ka0, qfA0, sA0); sA0 = wmma_bf16(ka1, qfA1, sA0);
        sA1 = wmma_bf16(kb0, qfA0, sA1); sA1 = wmma_bf16(kb1, qfA1, sA1);
        sB0 = wmma_bf16(ka0, qfB0, sB0); sB0 = wmma_bf16(ka1, qfB1, sB0);
        sB1 = wmma_bf16(kb0, qfB0, sB1); sB1 = wmma_bf16(kb1, qfB1, sB1);

        bfrag_t pfA = softmax_update(sA0 * scale, sA1 * scale, run_mA, run_sA, oA, hi);
        bfrag_t pfB = softmax_update(sB0 * scale, sB1 * scale, run_mB, run_sB, oB, hi);

        // V^T A-fragments (16 head dims x 32 keys each), shared by both tiles.
        #pragma unroll
        for (int t = 0; t < 4; ++t) {
            bfrag_t va = load_a_frag(vp, SEQ, 16 * t, ks, lane);
            oA[t] = wmma_bf16(va, pfA, oA[t]);
            oB[t] = wmma_bf16(va, pfB, oB[t]);
        }
    }

    float invA = 1.0f / run_sA;
    float invB = 1.0f / run_sB;
    #pragma unroll
    for (int t = 0; t < 4; ++t) { oA[t] = oA[t] * invA; oB[t] = oB[t] * invB; }

    // Write attn output bf16, row-major [b*2048+q][hq*64+d]
    int half = lane >> 4, lr = lane & 15;
    size_t rowA = ((size_t)(b * SEQ + q0 + lr))      * D_MODEL + (size_t)hq * D_HEAD;
    size_t rowB = ((size_t)(b * SEQ + q0 + 16 + lr)) * D_MODEL + (size_t)hq * D_HEAD;
    #pragma unroll
    for (int t = 0; t < 4; ++t) {
        #pragma unroll
        for (int v = 0; v < 8; ++v) {
            int d0 = 16 * t + v + half * 8;
            attnb[rowA + d0] = f2bf(oA[t][v]);
            attnb[rowB + d0] = f2bf(oB[t][v]);
        }
    }
}

// ---------------------------------------------------------------------------
// Output projection: out(4096 x 2048, fp32) = attn_bf @ Wo
// 32x64 macro-tile per wave, same blocking as qkv_gemm.
// ---------------------------------------------------------------------------
__global__ void __launch_bounds__(128, 1)
out_gemm_kernel(const unsigned short* __restrict__ attnb,
                const unsigned short* __restrict__ wot,
                float* __restrict__ out) {
    int lane = threadIdx.x & 31;
    int wave = threadIdx.x >> 5;
    int m0 = blockIdx.x * 32;                 // 128 row macro-tiles
    int n0 = (blockIdx.y * 4 + wave) * 64;    // 32 col macro-tiles

    v8f acc[2][4] = {};
    for (int k0 = 0; k0 < D_MODEL; k0 += 32) {
        bfrag_t bf[4];
        #pragma unroll
        for (int j = 0; j < 4; ++j)
            bf[j] = load_bt_frag(wot, D_MODEL, n0 + 16 * j, k0, lane);
        if (k0 + 64 <= D_MODEL) {
            __builtin_prefetch(attnb + (size_t)(m0 + (lane & 15)) * D_MODEL + k0 + 32, 0, 1);
            __builtin_prefetch(wot   + (size_t)(n0 + (lane & 15)) * D_MODEL + k0 + 32, 0, 1);
        }
        #pragma unroll
        for (int i = 0; i < 2; ++i) {
            bfrag_t af = load_a_frag(attnb, D_MODEL, m0 + 16 * i, k0, lane);
            #pragma unroll
            for (int j = 0; j < 4; ++j)
                acc[i][j] = wmma_bf16(af, bf[j], acc[i][j]);
        }
    }

    int half = lane >> 4, lr = lane & 15;
    #pragma unroll
    for (int i = 0; i < 2; ++i) {
        #pragma unroll
        for (int j = 0; j < 4; ++j) {
            int col = n0 + 16 * j + lr;
            #pragma unroll
            for (int v = 0; v < 8; ++v) {
                int row = m0 + 16 * i + v + half * 8;
                out[(size_t)row * D_MODEL + col] = acc[i][j][v];
            }
        }
    }
}

// ---------------------------------------------------------------------------
// Launch
// ---------------------------------------------------------------------------
extern "C" void kernel_launch(void* const* d_in, const int* in_sizes, int n_in,
                              void* d_out, int out_size, void* d_ws, size_t ws_size,
                              hipStream_t stream) {
    const float* x  = (const float*)d_in[0];
    const float* wq = (const float*)d_in[1];
    const float* wk = (const float*)d_in[2];
    const float* wv = (const float*)d_in[3];
    const float* wo = (const float*)d_in[4];
    float* out = (float*)d_out;

    // Workspace layout (bf16 buffers), ~60 MiB total.
    char* ws = (char*)d_ws;
    size_t off = 0;
    unsigned short* xb    = (unsigned short*)(ws + off); off += (size_t)ROWS * D_MODEL * 2;   // 16 MiB
    unsigned short* wt    = (unsigned short*)(ws + off); off += (size_t)N_QKV * D_MODEL * 2;  // 12 MiB
    unsigned short* wot   = (unsigned short*)(ws + off); off += (size_t)D_MODEL * D_MODEL * 2;// 8 MiB
    unsigned short* qbuf  = (unsigned short*)(ws + off); off += (size_t)BATCH * H_Q  * SEQ * D_HEAD * 2; // 16 MiB
    unsigned short* kbuf  = (unsigned short*)(ws + off); off += (size_t)BATCH * H_KV * SEQ * D_HEAD * 2; // 4 MiB
    unsigned short* vtbuf = (unsigned short*)(ws + off); off += (size_t)BATCH * H_KV * SEQ * D_HEAD * 2; // 4 MiB
    unsigned short* attnb = xb;  // x_bf dead after QKV GEMM; reuse for attention output

    // 1) fp32 -> bf16 conversions (weights transposed)
    {
        int n = ROWS * D_MODEL;
        cvt_x_kernel<<<(n + 255) / 256, 256, 0, stream>>>(x, xb, n);
    }
    {
        int n = N_QKV * D_MODEL;
        cvt_wqkv_kernel<<<(n + 255) / 256, 256, 0, stream>>>(wq, wk, wv, wt);
    }
    {
        int n = D_MODEL * D_MODEL;
        cvt_wo_kernel<<<(n + 255) / 256, 256, 0, stream>>>(wo, wot);
    }

    // 2) Fused QKV projection (4096 x 3072): 128 x 12 blocks of 4 waves
    qkv_gemm_kernel<<<dim3(ROWS / 32, N_QKV / 64 / 4), 128, 0, stream>>>(
        xb, wt, qbuf, kbuf, vtbuf);

    // 3) Flash attention: 16 x 64 blocks of 4 waves (32 queries per wave)
    attn_kernel<<<dim3(SEQ / 32 / 4, BATCH * H_Q), 128, 0, stream>>>(
        qbuf, kbuf, vtbuf, attnb);

    // 4) Output projection (4096 x 2048), fp32 result: 128 x 8 blocks
    out_gemm_kernel<<<dim3(ROWS / 32, D_MODEL / 64 / 4), 128, 0, stream>>>(
        attnb, wot, out);
}
